// MultiHeadSelfAttention_47725676593491
// MI455X (gfx1250) — matrix operand order repaired and checked by reading, not compiled, not run on hardware
//
#include <hip/hip_runtime.h>
#include <hip/hip_bf16.h>

#define BATCH   4
#define SEQ     2048
#define E_DIM   1024
#define N_HEADS 16
#define H_DIM   64

typedef __attribute__((ext_vector_type(16))) __bf16       v16bf;
typedef __attribute__((ext_vector_type(8)))  float        v8f;
typedef __attribute__((ext_vector_type(8)))  unsigned int v8u;
typedef __attribute__((ext_vector_type(4)))  unsigned int v4u;
typedef __attribute__((ext_vector_type(8)))  int          v8i;
typedef __attribute__((ext_vector_type(4)))  int          v4i;

static __device__ __forceinline__ unsigned short f2bf(float f) {
  unsigned int u = __float_as_uint(f);
  unsigned int r = 0x7FFFu + ((u >> 16) & 1u);        // round-to-nearest-even
  return (unsigned short)((u + r) >> 16);
}
static __device__ __forceinline__ unsigned int pack2bf(float lo, float hi) {
  return (unsigned int)f2bf(lo) | ((unsigned int)f2bf(hi) << 16);
}
static __device__ __forceinline__ v16bf as_v16bf(v8u u) {
  return __builtin_bit_cast(v16bf, u);
}
static __device__ __forceinline__ v8f wmma_bf16(v16bf a, v16bf b, v8f c) {
  // D = A(16x32) * B(32x16) + C, f32 accum
  return __builtin_amdgcn_wmma_f32_16x16x32_bf16(false, a, false, b, (short)0, c, false, false);
}

// TDM: async DMA of a 2D bf16 tile (tile1 rows x tile0 elems, row stride stride0 elems)
// from global to LDS at byte offset lds_off. pad!=0 appends 1 DWORD of LDS padding
// after every 16 DWORDs (i.e. per 32-elem row) -> [rows][17] u32 layout.
static __device__ __forceinline__ void tdm_load_2d_bf16(unsigned int lds_off,
                                                        const void* gptr,
                                                        unsigned int tensor0, unsigned int tensor1,
                                                        unsigned int tile0, unsigned int tile1,
                                                        unsigned int stride0, int pad) {
  const unsigned long long ga = (unsigned long long)(size_t)gptr;
  v4u g0;
  g0[0] = 1u;                                            // count=1, user mode
  g0[1] = lds_off;                                       // D#.lds_addr
  g0[2] = (unsigned int)(ga & 0xFFFFFFFFu);              // global_addr[31:0]
  g0[3] = (unsigned int)((ga >> 32) & 0x01FFFFFFu) | (2u << 30);  // addr[56:32], type=2
  unsigned int w0 = (1u << 16);                          // data_size=1 (2 bytes)
  if (pad) w0 |= (1u << 20) | (3u << 22);                // pad_enable, interval=16DW, amount=1DW
  v8i g1;
  g1[0] = (int)w0;
  g1[1] = (int)((tensor0 & 0xFFFFu) << 16);              // tensor_dim0[15:0]
  g1[2] = (int)(((tensor0 >> 16) & 0xFFFFu) | ((tensor1 & 0xFFFFu) << 16));
  g1[3] = (int)(((tensor1 >> 16) & 0xFFFFu) | ((tile0 & 0xFFFFu) << 16)); // tile_dim0
  g1[4] = (int)(tile1 & 0xFFFFu);                        // tile_dim1 (tile_dim2=0)
  g1[5] = (int)stride0;                                  // tensor_dim0_stride[31:0]
  g1[6] = 0;                                             // stride0 hi / stride1 lo
  g1[7] = 0;                                             // stride1 hi
  const v4i z4 = {0, 0, 0, 0};
  const v8i z8 = {0, 0, 0, 0, 0, 0, 0, 0};
  // 6-arg form (clang-23 / therock-10.0): groups 2/3 unused (2D tensor), trailing group zeroed
  __builtin_amdgcn_tensor_load_to_lds(g0, g1, z4, z4, z8, 0);
}

// A-operand loader for 16x32 bf16 tile from row-major A (lda elems/row).
// Lane m = lane&15; lo lanes hold K {0..7,16..23}, hi lanes {8..15,24..31}.
static __device__ __forceinline__ v16bf load_a_tile(const unsigned short* A, int lda,
                                                    int m0, int k0, int lane) {
  const int m  = m0 + (lane & 15);
  const int kb = k0 + ((lane >> 4) << 3);
  const unsigned int* row = (const unsigned int*)(A + (size_t)m * lda);
  const int p = kb >> 1;
  v8u u;
  u[0] = row[p + 0]; u[1] = row[p + 1]; u[2] = row[p + 2]; u[3] = row[p + 3];
  u[4] = row[p + 8]; u[5] = row[p + 9]; u[6] = row[p + 10]; u[7] = row[p + 11];
  return as_v16bf(u);
}

// ---------------- fp32 -> bf16 converts ----------------
__global__ __launch_bounds__(256)
void cvt_bf16_kernel(const float* __restrict__ in, unsigned short* __restrict__ out, int n) {
  for (int i = blockIdx.x * blockDim.x + threadIdx.x; i < n; i += gridDim.x * blockDim.x)
    out[i] = f2bf(in[i]);
}

// W (K,N) fp32 -> Wt (N,K) bf16 so B-operand K-pairs are contiguous
__global__ __launch_bounds__(256)
void cvt_transpose_kernel(const float* __restrict__ W, unsigned short* __restrict__ Wt,
                          int K, int N) {
  const int total = K * N;
  for (int i = blockIdx.x * blockDim.x + threadIdx.x; i < total; i += gridDim.x * blockDim.x) {
    const int k = i / N, n = i - k * N;
    Wt[(size_t)n * K + k] = f2bf(W[i]);
  }
}

// ---------------- GEMM: C(M,N) = A(M,K)bf16 * Bt(N,K)bf16 + bias ----------------
// 128 threads = 4 waves; wave w computes rows [m0+32w, +32) x cols [n0, n0+64).
// B tile (64x32) staged by the Tensor Data Mover into double-buffered LDS.
// mode 0: plain fp32 output. mode 1: QKV epilogue scattering bf16 Q/K/V^T.
__global__ __launch_bounds__(128)
void gemm_bf16_kernel(const unsigned short* __restrict__ A,
                      const unsigned short* __restrict__ Bt,
                      const float* __restrict__ bias,
                      float* __restrict__ C,
                      unsigned short* __restrict__ Qw,
                      unsigned short* __restrict__ Kw,
                      unsigned short* __restrict__ Vtw,
                      int M, int N, int Kdim, int mode) {
  __shared__ unsigned int Bs[2][64][17];     // TDM fills 64x(16 u32) rows + 1 u32 pad/row
  const int tid  = threadIdx.x;
  const int lane = tid & 31;
  const int wave = tid >> 5;
  const int m0 = blockIdx.x * 128 + wave * 32;
  const int n0 = blockIdx.y * 64;

  const v8f vzero = {0.f, 0.f, 0.f, 0.f, 0.f, 0.f, 0.f, 0.f};
  v8f acc[2][4];
#pragma unroll
  for (int rt = 0; rt < 2; ++rt)
#pragma unroll
    for (int i = 0; i < 4; ++i) acc[rt][i] = vzero;

  const int half8 = (lane >> 4) << 3;        // B-operand K base: 0 (K 0..7,16..23) / 8 (K 8..15,24..31)
  const unsigned int bs_lds = (unsigned int)(size_t)&Bs[0][0][0];
  const unsigned int buf_bytes = 64u * 17u * 4u;
  const unsigned short* bt_tile = Bt + (size_t)n0 * Kdim;

  if (wave == 0)   // preload first B tile (k0 = 0)
    tdm_load_2d_bf16(bs_lds, bt_tile, (unsigned)Kdim, (unsigned)N, 32u, 64u, (unsigned)Kdim, 1);

  int cur = 0;
  for (int k0 = 0; k0 < Kdim; k0 += 32) {
    if (wave == 0) __builtin_amdgcn_s_wait_tensorcnt(0);   // buf[cur] DMA complete
    __syncthreads();                                       // visible to all; buf[cur^1] free
    if (wave == 0 && (k0 + 32) < Kdim)
      tdm_load_2d_bf16(bs_lds + (unsigned)(cur ^ 1) * buf_bytes, bt_tile + k0 + 32,
                       (unsigned)Kdim, (unsigned)N, 32u, 64u, (unsigned)Kdim, 1);

    if (k0 + 64 < Kdim)                                    // speculative next-A prefetch
      __builtin_prefetch(A + (size_t)(m0 + (lane & 15)) * Kdim + k0 + 64, 0, 1);

    const v16bf a0 = load_a_tile(A, Kdim, m0,      k0, lane);
    const v16bf a1 = load_a_tile(A, Kdim, m0 + 16, k0, lane);
#pragma unroll
    for (int sub = 0; sub < 4; ++sub) {
      const int nloc = sub * 16 + (lane & 15);
      v8u u;
#pragma unroll
      for (int v = 0; v < 8; ++v) u[v] = Bs[cur][nloc][(half8 >> 1) + (v & 3) + ((v >> 2) << 3)];
      const v16bf b = as_v16bf(u);
      acc[0][sub] = wmma_bf16(a0, b, acc[0][sub]);
      acc[1][sub] = wmma_bf16(a1, b, acc[1][sub]);
    }
    cur ^= 1;
  }

  const int nl = lane & 15;
  const int mh = (lane >> 4) << 3;
#pragma unroll
  for (int rt = 0; rt < 2; ++rt) {
#pragma unroll
    for (int sub = 0; sub < 4; ++sub) {
      const int ncol = n0 + sub * 16 + nl;
      const float bv = bias[ncol];
#pragma unroll
      for (int r = 0; r < 8; ++r) {
        const int mrow = m0 + rt * 16 + mh + r;
        const float val = acc[rt][sub][r] + bv;
        if (mode == 0) {
          C[(size_t)mrow * N + ncol] = val;
        } else {
          // qkv column layout per head: [q(64) | k(64) | v(64)]
          const int h = ncol / 192;
          const int w = ncol - h * 192;
          const int t = w >> 6;
          const int d = w & 63;
          const int bb = mrow >> 11;              // / SEQ
          const int s  = mrow & (SEQ - 1);
          const unsigned short bf = f2bf(val);
          if (t == 0)      Qw[((((size_t)bb * N_HEADS + h) * SEQ + s) << 6) + d] = bf;
          else if (t == 1) Kw[((((size_t)bb * N_HEADS + h) * SEQ + s) << 6) + d] = bf;
          else             Vtw[((((size_t)bb * N_HEADS + h) << 6) + d) * SEQ + s] = bf; // (hd,S)
        }
      }
    }
  }
}

// ---------------- causal flash attention: one wave per 16-row Q tile ----------------
__global__ __launch_bounds__(256)
void attn_kernel(const unsigned short* __restrict__ Q,
                 const unsigned short* __restrict__ Kc,
                 const unsigned short* __restrict__ Vt,
                 unsigned short* __restrict__ O) {
  __shared__ float Pt[8][16][33];          // per-wave C-layout -> A-layout transpose scratch
  const int tid  = threadIdx.x;
  const int lane = tid & 31;
  const int wave = tid >> 5;
  const int qt = blockIdx.x * 8 + wave;    // global q-tile id over B*H*(S/16)
  const int bh = qt >> 7;                  // 128 q-tiles per (b,h)
  const int q0 = (qt & 127) << 4;
  const int b  = bh >> 4;
  const int h  = bh & 15;

  const unsigned short* Qp = Q  + (size_t)bh * SEQ * H_DIM;
  const unsigned short* Kp = Kc + (size_t)bh * SEQ * H_DIM;
  const unsigned short* Vp = Vt + (size_t)bh * H_DIM * SEQ;   // (hd, S)

  const int nl    = lane & 15;
  const int half  = lane >> 4;
  const int half8 = half << 3;

  const v16bf aq0 = load_a_tile(Qp, H_DIM, q0, 0, lane);   // d = 0..31
  const v16bf aq1 = load_a_tile(Qp, H_DIM, q0, 32, lane);  // d = 32..63

  const v8f vzero = {0.f, 0.f, 0.f, 0.f, 0.f, 0.f, 0.f, 0.f};
  v8f o[4];
#pragma unroll
  for (int i = 0; i < 4; ++i) o[i] = vzero;
  float rmax[8], rsum[8];
#pragma unroll
  for (int r = 0; r < 8; ++r) { rmax[r] = -1e30f; rsum[r] = 0.f; }

  const int kend = q0 + 16;                // causal: cols <= q0+15
  for (int k0 = 0; k0 < kend; k0 += 32) {
    v8f s[2];
#pragma unroll
    for (int j = 0; j < 2; ++j) {
      const int kc = k0 + j * 16;
      const unsigned int* krow = (const unsigned int*)(Kp + (size_t)(kc + nl) * H_DIM);
      v8f sc = vzero;
      v8u u0, u1;
#pragma unroll
      for (int v = 0; v < 8; ++v) u0[v] = krow[(half8 >> 1) + (v & 3) + ((v >> 2) << 3)];        // d 0..31
      sc = wmma_bf16(aq0, as_v16bf(u0), sc);
#pragma unroll
      for (int v = 0; v < 8; ++v) u1[v] = krow[16 + (half8 >> 1) + (v & 3) + ((v >> 2) << 3)];   // d 32..63
      sc = wmma_bf16(aq1, as_v16bf(u1), sc);
#pragma unroll
      for (int r = 0; r < 8; ++r) {                               // scale + causal mask
        const int qrow = q0 + half8 + r;
        const int kcol = kc + nl;
        s[j][r] = (kcol <= qrow) ? sc[r] * 0.125f : -1e30f;
      }
    }
    // online softmax: per-row stats via xor-shuffle over the 16-lane group
#pragma unroll
    for (int r = 0; r < 8; ++r) {
      float mloc = fmaxf(s[0][r], s[1][r]);
#pragma unroll
      for (int off = 1; off < 16; off <<= 1) mloc = fmaxf(mloc, __shfl_xor(mloc, off, 32));
      const float mnew = fmaxf(rmax[r], mloc);
      const float corr = __expf(rmax[r] - mnew);
      const float p0 = __expf(s[0][r] - mnew);
      const float p1 = __expf(s[1][r] - mnew);
      float ps = p0 + p1;
#pragma unroll
      for (int off = 1; off < 16; off <<= 1) ps += __shfl_xor(ps, off, 32);
      rsum[r] = rsum[r] * corr + ps;
      rmax[r] = mnew;
      s[0][r] = p0; s[1][r] = p1;
#pragma unroll
      for (int sub = 0; sub < 4; ++sub) o[sub][r] *= corr;
    }
    // transpose P(16x32) from C-layout to A-layout through per-wave LDS
#pragma unroll
    for (int r = 0; r < 8; ++r) {
      Pt[wave][half8 + r][nl]      = s[0][r];
      Pt[wave][half8 + r][16 + nl] = s[1][r];
    }
    asm volatile("s_wait_dscnt 0" ::: "memory");   // wave-internal LDS WAR/RAW ordering
    v8u pu;
    {
      const float* prow = &Pt[wave][nl][0];
#pragma unroll
      for (int v = 0; v < 8; ++v) {
        const int kk = (v < 4) ? (half8 + 2 * v) : (half8 + 16 + 2 * (v - 4));
        pu[v] = pack2bf(prow[kk], prow[kk + 1]);
      }
    }
    const v16bf ap = as_v16bf(pu);
    // O += P(16x32) * V(32x16) per d-subtile; V^T rows give contiguous K-pairs
#pragma unroll
    for (int sub = 0; sub < 4; ++sub) {
      const unsigned int* vrow = (const unsigned int*)(Vp + (size_t)(sub * 16 + nl) * SEQ);
      v8u u;
#pragma unroll
      for (int v = 0; v < 8; ++v) u[v] = vrow[(k0 >> 1) + (half8 >> 1) + (v & 3) + ((v >> 2) << 3)];
      o[sub] = wmma_bf16(ap, as_v16bf(u), o[sub]);
    }
  }

#pragma unroll
  for (int r = 0; r < 8; ++r) rsum[r] = 1.f / rsum[r];
#pragma unroll
  for (int sub = 0; sub < 4; ++sub) {
#pragma unroll
    for (int r = 0; r < 8; ++r) {
      const float val = o[sub][r] * rsum[r];
      const int srow = q0 + half8 + r;
      const int d    = sub * 16 + nl;
      O[(size_t)(b * SEQ + srow) * E_DIM + h * H_DIM + d] = f2bf(val);  // (B,S,E) bf16
    }
  }
}

extern "C" void kernel_launch(void* const* d_in, const int* in_sizes, int n_in,
                              void* d_out, int out_size, void* d_ws, size_t ws_size,
                              hipStream_t stream) {
  const float* x      = (const float*)d_in[0];
  const float* W_qkv  = (const float*)d_in[1];
  const float* b_qkv  = (const float*)d_in[2];
  const float* W_out  = (const float*)d_in[3];
  const float* b_out  = (const float*)d_in[4];
  float* out = (float*)d_out;

  char* ws = (char*)d_ws;
  unsigned short* xb    = (unsigned short*)(ws + ((size_t)0  << 20));  // 16 MiB
  unsigned short* wqkvt = (unsigned short*)(ws + ((size_t)16 << 20));  //  6 MiB
  unsigned short* woutt = (unsigned short*)(ws + ((size_t)22 << 20));  //  2 MiB
  unsigned short* qw    = (unsigned short*)(ws + ((size_t)24 << 20));  // 16 MiB
  unsigned short* kw    = (unsigned short*)(ws + ((size_t)40 << 20));  // 16 MiB
  unsigned short* vtw   = (unsigned short*)(ws + ((size_t)56 << 20));  // 16 MiB
  unsigned short* attn  = (unsigned short*)(ws + ((size_t)72 << 20));  // 16 MiB

  const int Mrows = BATCH * SEQ;  // 8192

  cvt_bf16_kernel<<<4096, 256, 0, stream>>>(x, xb, Mrows * E_DIM);
  cvt_transpose_kernel<<<4096, 256, 0, stream>>>(W_qkv, wqkvt, E_DIM, 3 * E_DIM);
  cvt_transpose_kernel<<<2048, 256, 0, stream>>>(W_out, woutt, E_DIM, E_DIM);

  dim3 g1(Mrows / 128, (3 * E_DIM) / 64);
  gemm_bf16_kernel<<<g1, 128, 0, stream>>>(xb, wqkvt, b_qkv, nullptr, qw, kw, vtw,
                                           Mrows, 3 * E_DIM, E_DIM, 1);

  attn_kernel<<<(BATCH * N_HEADS * (SEQ / 16)) / 8, 256, 0, stream>>>(qw, kw, vtw, attn);

  dim3 g2(Mrows / 128, E_DIM / 64);
  gemm_bf16_kernel<<<g2, 128, 0, stream>>>(attn, woutt, b_out, out, nullptr, nullptr, nullptr,
                                           Mrows, E_DIM, E_DIM, 0);
}
